// CogvlmVisionAttention_67138928771191
// MI455X (gfx1250) — matrix operand order repaired and checked by reading, not compiled, not run on hardware
//
#include <hip/hip_runtime.h>
#include <hip/hip_bf16.h>

// ---------------- problem constants ----------------
#define Bc   8
#define Sc   1226
#define Hc   1792
#define NHc  16
#define HDc  112
#define K3c  5376            // 3*H
#define SPc  1280            // S padded: multiple of 32, and 80 q-tiles = 4 waves * 20 blocks
#define HDPc 128             // HD padded to multiple of 32
#define MROWS 9808           // B*S
#define MT32  307            // ceil(9808/32)
#define NTILES (SPc / 32)    // 40 t-steps in attention
#define SCALEc 0.09449111825230679f  // 1/sqrt(112)

// ---------------- types ----------------
typedef __bf16 bf16_t;
typedef __attribute__((ext_vector_type(16))) __bf16 v16bf;
typedef __attribute__((ext_vector_type(8)))  __bf16 v8bf;
typedef __attribute__((ext_vector_type(8)))  float  v8f;
typedef int v4i_t __attribute__((vector_size(16)));

union V16 { v16bf v; v8bf h[2]; };

static __device__ inline v8f wmma_bf16(v16bf a, v16bf b, v8f c) {
  return __builtin_amdgcn_wmma_f32_16x16x32_bf16(false, a, false, b,
                                                 (short)0, c, false, false);
}

static __device__ inline bf16_t f2bf(float f) {
  unsigned u = __builtin_bit_cast(unsigned, f);
  unsigned r = u + 0x7FFFu + ((u >> 16) & 1u);   // round-to-nearest-even
  unsigned short h = (unsigned short)(r >> 16);
  return __builtin_bit_cast(bf16_t, h);
}

// ---------------- gfx1250 async global->LDS copy (with sync fallback) ----------------
#if defined(__has_builtin)
#  if __has_builtin(__builtin_amdgcn_global_load_async_to_lds_b128)
#    define HAS_ASYNC 1
#  endif
#endif
#ifndef HAS_ASYNC
#  define HAS_ASYNC 0
#endif

static __device__ inline void cp16(const char* g, char* l) {
#if HAS_ASYNC
  __builtin_amdgcn_global_load_async_to_lds_b128(
      (__attribute__((address_space(1))) v4i_t*)(g),
      (__attribute__((address_space(3))) v4i_t*)(l), 0, 0);
#else
  *(uint4*)l = *(const uint4*)g;
#endif
}

static __device__ inline void cp_wait_all() {
#if HAS_ASYNC
#  if __has_builtin(__builtin_amdgcn_s_wait_asynccnt)
  __builtin_amdgcn_s_wait_asynccnt(0);
#  else
  asm volatile("s_wait_asynccnt 0x0" ::: "memory");
#  endif
#endif
}

// ---------------- workspace layout (bytes, 256-aligned) ----------------
#define QKVW_BYTES ((size_t)K3c * Hc * 2)
#define DW_BYTES   ((size_t)Hc * Hc * 2)
#define QBUF_BYTES ((size_t)Bc * NHc * SPc * HDPc * 2)   // 41,943,040 each
#define O_QKVW 0
#define O_DW   (O_QKVW + QKVW_BYTES)
#define O_Q    (O_DW + DW_BYTES)
#define O_K    (O_Q + QBUF_BYTES)
#define O_VT   (O_K + QBUF_BYTES)
#define O_ATTN (O_VT + QBUF_BYTES)

// ---------------- kernel 0: zero padded Q/K/Vt region ----------------
__global__ void zero_ws_kernel(uint4* p, unsigned n16) {
  unsigned i = blockIdx.x * 256u + threadIdx.x;
  if (i < n16) p[i] = make_uint4(0u, 0u, 0u, 0u);
}

// ---------------- kernel 1: fp32 [K,N] -> bf16 [N,K] transpose ----------------
__global__ void cvt_transpose_kernel(const float* __restrict__ src,
                                     bf16_t* __restrict__ dst,
                                     int K, int N) {
  unsigned idx = blockIdx.x * 256u + threadIdx.x;
  if (idx >= (unsigned)(N * K)) return;
  int n = idx / K;
  int k = idx - n * K;
  dst[idx] = f2bf(src[(size_t)k * N + n]);
}

// ---------------- shared 32x64 WMMA GEMM main loop ----------------
// A: [M,K] row-major (fp32 or bf16); Wt: [N,K] bf16 row-major (pre-transposed).
// Computes a 32(M) x 64(N) tile per wave: 8 accumulators, 8 WMMA per k-step.
template <bool AF32>
static __device__ inline void gemm32x64(const void* __restrict__ Ap,
                                        const bf16_t* __restrict__ Wt,
                                        int K, int r0, int r1, int n0, v8f* acc) {
  const int lane  = threadIdx.x & 31;
  const int ml    = lane & 15;
  const int hi    = lane >> 4;
  const int g     = hi * 16;    // A-fragment K half
  const int koffB = hi * 16;    // B-fragment K half

  for (int k0 = 0; k0 < K; k0 += 32) {
    V16 a0, a1;
    if constexpr (AF32) {
      const float* A0 = (const float*)Ap + (size_t)r0 * K + k0;
      const float* A1 = (const float*)Ap + (size_t)r1 * K + k0;
      float4 x0 = *(const float4*)(A0 + g);
      float4 x1 = *(const float4*)(A0 + g + 4);
      float4 y0 = *(const float4*)(A0 + g + 8);
      float4 y1 = *(const float4*)(A0 + g + 12);
      float4 z0 = *(const float4*)(A1 + g);
      float4 z1 = *(const float4*)(A1 + g + 4);
      float4 w0 = *(const float4*)(A1 + g + 8);
      float4 w1 = *(const float4*)(A1 + g + 12);
      float f0[16] = {x0.x, x0.y, x0.z, x0.w, x1.x, x1.y, x1.z, x1.w,
                      y0.x, y0.y, y0.z, y0.w, y1.x, y1.y, y1.z, y1.w};
      float f1[16] = {z0.x, z0.y, z0.z, z0.w, z1.x, z1.y, z1.z, z1.w,
                      w0.x, w0.y, w0.z, w0.w, w1.x, w1.y, w1.z, w1.w};
#pragma unroll
      for (int i = 0; i < 16; ++i) { a0.v[i] = f2bf(f0[i]); a1.v[i] = f2bf(f1[i]); }
      if (k0 + 32 < K) { __builtin_prefetch(A0 + 32, 0, 1); __builtin_prefetch(A1 + 32, 0, 1); }
    } else {
      const bf16_t* A0 = (const bf16_t*)Ap + (size_t)r0 * K + k0;
      const bf16_t* A1 = (const bf16_t*)Ap + (size_t)r1 * K + k0;
      a0.h[0] = *(const v8bf*)(A0 + g);
      a0.h[1] = *(const v8bf*)(A0 + g + 8);
      a1.h[0] = *(const v8bf*)(A1 + g);
      a1.h[1] = *(const v8bf*)(A1 + g + 8);
      if (k0 + 32 < K) { __builtin_prefetch(A0 + 32, 0, 1); __builtin_prefetch(A1 + 32, 0, 1); }
    }
#pragma unroll
    for (int t = 0; t < 4; ++t) {
      const bf16_t* bp = Wt + (size_t)(n0 + t * 16 + ml) * K + k0 + koffB;
      v16bf b = *(const v16bf*)bp;
      acc[t]     = wmma_bf16(a0.v, b, acc[t]);
      acc[4 + t] = wmma_bf16(a1.v, b, acc[4 + t]);
    }
  }
}

// ---------------- kernel 2: QKV GEMM + bias + scatter to Q/K/Vt ----------------
__global__ __launch_bounds__(256)
void qkv_gemm_kernel(const float* __restrict__ hs,
                     const bf16_t* __restrict__ Wt,
                     const float* __restrict__ bias,
                     bf16_t* __restrict__ Qb, bf16_t* __restrict__ Kb,
                     bf16_t* __restrict__ Vt) {
  const int wave  = threadIdx.x >> 5;
  const int mtile = blockIdx.y * 8 + wave;
  if (mtile >= MT32) return;
  const int n0 = blockIdx.x * 64;
  const int m0 = mtile * 32;

  const int lane = threadIdx.x & 31;
  const int ml = lane & 15, hi = lane >> 4;
  const int r0 = min(m0 + ml, MROWS - 1);          // clamp tail rows (garbage, not stored)
  const int r1 = min(m0 + 16 + ml, MROWS - 1);

  v8f acc[8] = {};
  gemm32x64<true>(hs, Wt, Hc, r0, r1, n0, acc);

#pragma unroll
  for (int half = 0; half < 2; ++half) {
#pragma unroll
    for (int t = 0; t < 4; ++t) {
      int n = n0 + t * 16 + ml;
      float bv = bias[n];
      int sec = n / Hc;
      int r   = n - sec * Hc;
      int hh  = r / HDc;
      int d   = r - hh * HDc;
#pragma unroll
      for (int j = 0; j < 8; ++j) {
        int m = m0 + half * 16 + hi * 8 + j;
        if (m >= MROWS) continue;
        int b = m / Sc;
        int s = m - b * Sc;
        int bh = b * NHc + hh;
        float v = acc[half * 4 + t][j] + bv;
        if (sec == 0)      Qb[((size_t)bh * SPc + s) * HDPc + d] = f2bf(v * SCALEc);
        else if (sec == 1) Kb[((size_t)bh * SPc + s) * HDPc + d] = f2bf(v);
        else               Vt[((size_t)bh * HDPc + d) * SPc + s] = f2bf(v);
      }
    }
  }
}

// ---------------- kernel 3: flash attention ----------------
// 4 waves per block, each owns 16 query rows; K/V tiles double-buffered in LDS
// via async global->LDS copies, shared by all 4 waves.
__global__ __launch_bounds__(128)
void attn_kernel(const bf16_t* __restrict__ Qb, const bf16_t* __restrict__ Kb,
                 const bf16_t* __restrict__ Vt, bf16_t* __restrict__ attnB) {
  __shared__ alignas(32) bf16_t Ktile[2][32][128];   // [buf][t][d]   8KB each
  __shared__ alignas(32) bf16_t Vtile[2][128][32];   // [buf][d][t]   8KB each
  __shared__ alignas(32) bf16_t Pst[4][16][40];      // per-wave P staging

  const int bh = blockIdx.y;
  const int b  = bh >> 4;
  const int h  = bh & 15;
  const int w    = threadIdx.x >> 5;                 // wave id 0..3
  const int lane = threadIdx.x & 31;
  const int tid  = threadIdx.x;
  const int m0 = (blockIdx.x * 4 + w) * 16;          // 16 query rows per wave
  const int ml = lane & 15, hi = lane >> 4;
  const int g = hi * 16, koffB = hi * 16;

  const bf16_t* Qh = Qb + (size_t)bh * SPc * HDPc;
  const bf16_t* Kh = Kb + (size_t)bh * SPc * HDPc;
  const bf16_t* Vh = Vt + (size_t)bh * HDPc * SPc;

  // Q fragments for all 4 K-chunks of the padded head dim (128 = 4*32)
  V16 qf[4];
#pragma unroll
  for (int kc = 0; kc < 4; ++kc) {
    const bf16_t* qp = Qh + (size_t)(m0 + ml) * HDPc + kc * 32;
    qf[kc].h[0] = *(const v8bf*)(qp + g);
    qf[kc].h[1] = *(const v8bf*)(qp + g + 8);
  }

  v8f O[8] = {};
  float rmax[8], rsum[8];
#pragma unroll
  for (int j = 0; j < 8; ++j) { rmax[j] = -__builtin_inff(); rsum[j] = 0.0f; }

  // cooperative tile copy: K tile is 8KB contiguous; V tile is 128 rows x 64B
  auto copy_tiles = [&](int it, int bufn) {
    const int t0 = it * 32;
    const char* Kg = (const char*)Kh + (size_t)t0 * HDPc * 2;
    char* Kl = (char*)&Ktile[bufn][0][0];
#pragma unroll
    for (int i = 0; i < 4; ++i) {
      int off = (tid + i * 128) * 16;
      cp16(Kg + off, Kl + off);
    }
    char* Vl = (char*)&Vtile[bufn][0][0];
#pragma unroll
    for (int i = 0; i < 4; ++i) {
      int off = (tid + i * 128) * 16;
      int d = off >> 6;        // row (64B per row)
      int c = off & 63;        // byte within row
      cp16((const char*)Vh + ((size_t)d * SPc + t0) * 2 + c, Vl + off);
    }
  };

  copy_tiles(0, 0);

  for (int it = 0; it < NTILES; ++it) {
    const int buf = it & 1;
    const int t0  = it * 32;

    cp_wait_all();        // this wave's async loads for tile `it` have landed
    __syncthreads();      // all waves: tile `it` visible; prev compute finished
    if (it + 1 < NTILES) copy_tiles(it + 1, buf ^ 1);   // overlap next DMA

    // ---- scores: 16x32 tile = two 16x16 fragments, K from LDS ----
    v8f s0 = {}, s1 = {};
#pragma unroll
    for (int kc = 0; kc < 4; ++kc) {
      v16bf k0v = *(const v16bf*)&Ktile[buf][ml][kc * 32 + koffB];
      s0 = wmma_bf16(qf[kc].v, k0v, s0);
      v16bf k1v = *(const v16bf*)&Ktile[buf][16 + ml][kc * 32 + koffB];
      s1 = wmma_bf16(qf[kc].v, k1v, s1);
    }
    // ---- mask padded keys ----
    if (t0 + ml >= Sc) {
#pragma unroll
      for (int j = 0; j < 8; ++j) s0[j] = -1.0e30f;
    }
    if (t0 + 16 + ml >= Sc) {
#pragma unroll
      for (int j = 0; j < 8; ++j) s1[j] = -1.0e30f;
    }
    // ---- online softmax ----
    float alpha[8];
#pragma unroll
    for (int j = 0; j < 8; ++j) {
      float mx = fmaxf(s0[j], s1[j]);
      mx = fmaxf(mx, __shfl_xor(mx, 1, 32));
      mx = fmaxf(mx, __shfl_xor(mx, 2, 32));
      mx = fmaxf(mx, __shfl_xor(mx, 4, 32));
      mx = fmaxf(mx, __shfl_xor(mx, 8, 32));
      float mnew = fmaxf(rmax[j], mx);
      alpha[j] = __expf(rmax[j] - mnew);
      rmax[j] = mnew;
      float p0 = __expf(s0[j] - mnew);
      float p1 = __expf(s1[j] - mnew);
      s0[j] = p0; s1[j] = p1;
      float rs = p0 + p1;
      rs += __shfl_xor(rs, 1, 32);
      rs += __shfl_xor(rs, 2, 32);
      rs += __shfl_xor(rs, 4, 32);
      rs += __shfl_xor(rs, 8, 32);
      rsum[j] = rsum[j] * alpha[j] + rs;
    }
#pragma unroll
    for (int dt = 0; dt < 8; ++dt)
#pragma unroll
      for (int j = 0; j < 8; ++j) O[dt][j] *= alpha[j];

    // ---- C-layout -> A-layout for P via per-wave LDS staging ----
#pragma unroll
    for (int j = 0; j < 8; ++j) {
      int r = hi * 8 + j;
      Pst[w][r][ml]      = f2bf(s0[j]);
      Pst[w][r][16 + ml] = f2bf(s1[j]);
    }
    V16 pf;
    pf.h[0] = *(const v8bf*)(&Pst[w][ml][g]);
    pf.h[1] = *(const v8bf*)(&Pst[w][ml][g + 8]);

    // ---- O += P * V, V tile from LDS (pre-transposed: contiguous over t) ----
#pragma unroll
    for (int dt = 0; dt < 8; ++dt) {
      v16bf vv = *(const v16bf*)&Vtile[buf][dt * 16 + ml][koffB];
      O[dt] = wmma_bf16(pf.v, vv, O[dt]);
    }
  }

  // ---- normalize + store bf16 rows for the dense GEMM ----
  float rinv[8];
#pragma unroll
  for (int j = 0; j < 8; ++j) rinv[j] = 1.0f / rsum[j];
#pragma unroll
  for (int dt = 0; dt < 7; ++dt) {   // dt==7 is the zero pad
    int d = dt * 16 + ml;
#pragma unroll
    for (int j = 0; j < 8; ++j) {
      int s = m0 + hi * 8 + j;
      if (s < Sc)
        attnB[(size_t)(b * Sc + s) * Hc + h * HDc + d] = f2bf(O[dt][j] * rinv[j]);
    }
  }
}

// ---------------- kernel 4: dense GEMM + bias -> fp32 out ----------------
__global__ __launch_bounds__(256)
void dense_gemm_kernel(const bf16_t* __restrict__ attnB,
                       const bf16_t* __restrict__ Wt,
                       const float* __restrict__ bias,
                       float* __restrict__ out) {
  const int wave  = threadIdx.x >> 5;
  const int mtile = blockIdx.y * 8 + wave;
  if (mtile >= MT32) return;
  const int n0 = blockIdx.x * 64;
  const int m0 = mtile * 32;

  const int lane = threadIdx.x & 31;
  const int ml = lane & 15, hi = lane >> 4;
  const int r0 = min(m0 + ml, MROWS - 1);
  const int r1 = min(m0 + 16 + ml, MROWS - 1);

  v8f acc[8] = {};
  gemm32x64<false>(attnB, Wt, Hc, r0, r1, n0, acc);

#pragma unroll
  for (int half = 0; half < 2; ++half) {
#pragma unroll
    for (int t = 0; t < 4; ++t) {
      int n = n0 + t * 16 + ml;
      float bv = bias[n];
#pragma unroll
      for (int j = 0; j < 8; ++j) {
        int m = m0 + half * 16 + hi * 8 + j;
        if (m < MROWS) out[(size_t)m * Hc + n] = acc[half * 4 + t][j] + bv;
      }
    }
  }
}

// ---------------- launcher ----------------
extern "C" void kernel_launch(void* const* d_in, const int* in_sizes, int n_in,
                              void* d_out, int out_size, void* d_ws, size_t ws_size,
                              hipStream_t stream) {
  (void)in_sizes; (void)n_in; (void)out_size; (void)ws_size;

  const float* hs      = (const float*)d_in[0];  // [B,S,H]
  const float* qkv_w   = (const float*)d_in[1];  // [H,3H]
  const float* qkv_b   = (const float*)d_in[2];  // [3H]
  const float* dense_w = (const float*)d_in[3];  // [H,H]
  const float* dense_b = (const float*)d_in[4];  // [H]
  float* out = (float*)d_out;                    // [B,S,H]

  char* ws = (char*)d_ws;
  bf16_t* qkvWt = (bf16_t*)(ws + O_QKVW);
  bf16_t* dWt   = (bf16_t*)(ws + O_DW);
  bf16_t* Qb    = (bf16_t*)(ws + O_Q);
  bf16_t* Kb    = (bf16_t*)(ws + O_K);
  bf16_t* Vt    = (bf16_t*)(ws + O_VT);
  bf16_t* attnB = (bf16_t*)(ws + O_ATTN);

  // 0) zero the padded Q/K/Vt region
  {
    unsigned n16 = (unsigned)((3 * QBUF_BYTES) / 16);
    zero_ws_kernel<<<dim3((n16 + 255u) / 256u), dim3(256), 0, stream>>>(
        (uint4*)(ws + O_Q), n16);
  }
  // 1) weight convert + transpose
  {
    unsigned tot = (unsigned)(K3c * Hc);
    cvt_transpose_kernel<<<dim3((tot + 255u) / 256u), dim3(256), 0, stream>>>(
        qkv_w, qkvWt, Hc, K3c);
    tot = (unsigned)(Hc * Hc);
    cvt_transpose_kernel<<<dim3((tot + 255u) / 256u), dim3(256), 0, stream>>>(
        dense_w, dWt, Hc, Hc);
  }
  // 2) QKV projection
  qkv_gemm_kernel<<<dim3(K3c / 64, (MT32 + 7) / 8), dim3(256), 0, stream>>>(
      hs, qkvWt, qkv_b, Qb, Kb, Vt);
  // 3) flash attention (4 waves/block share LDS K/V tiles)
  attn_kernel<<<dim3(SPc / 64, Bc * NHc), dim3(128), 0, stream>>>(Qb, Kb, Vt, attnB);
  // 4) output projection
  dense_gemm_kernel<<<dim3(Hc / 64, (MT32 + 7) / 8), dim3(256), 0, stream>>>(
      attnB, dWt, dense_b, out);
}
